// SAModule_3358664426063
// MI455X (gfx1250) — matrix operand order repaired and checked by reading, not compiled, not run on hardware
//
#include <hip/hip_runtime.h>

// ---------------------------------------------------------------------------
// PointNet++ SA module for MI455X (gfx1250, wave32, WMMA f16 16x16x32).
// Pipeline: prep -> FPS -> ball query -> [GEMM1 stats] -> finalize BN1 ->
//           [GEMM1+BN1+GEMM2 stats] -> finalize BN2 -> [full recompute + max]
// GEMMs are tiny (13 GFLOP total, weights L2-resident) so we recompute instead
// of spilling 134MB of activations to HBM.
// ---------------------------------------------------------------------------

#define Bc   8
#define Nc   2048
#define Mc   1024
#define Kc   64
#define Fc   64
#define C1c  64
#define C2c  128
#define R2c  0.04f      // R*R, R = 0.2
#define EPSc 1e-5f
#define NG   (Bc * Mc)  // 8192 groups
#define LDA  104        // f16 K-stride for hA / W1^T (67 -> pad 96, stride 104: bank-conflict free)
#define LDB  72         // f16 K-stride for hmid / W2^T (64 -> stride 72)

typedef __attribute__((ext_vector_type(16))) _Float16 v16h;
typedef __attribute__((ext_vector_type(8)))  _Float16 v8h;
typedef __attribute__((ext_vector_type(8)))  float    v8f;

// ---------------- WMMA fragment loaders (16-bit A 16x32 / B 32x16 layouts) --

// A (MxK): lane l(0-15) row M=l. VGPR0-3: K=k0+hi*8..+7, VGPR4-7: K=k0+16+hi*8..+7
__device__ __forceinline__ v16h load_frag_A(const _Float16* lds, int row0, int k0,
                                            int stride, int lane) {
  int l = lane & 15, hi = lane >> 4;
  const _Float16* p = lds + (row0 + l) * stride + k0 + hi * 8;
  v8h a = *(const v8h*)p;
  v8h b = *(const v8h*)(p + 16);
  return __builtin_shufflevector(a, b, 0,1,2,3,4,5,6,7,8,9,10,11,12,13,14,15);
}

// B (KxN) stored as W^T row-major [col][K]: lane l col N=l, K=k0+hi*8..+7 then k0+16+hi*8..+7
__device__ __forceinline__ v16h load_frag_B(const _Float16* ldsT, int col0, int k0,
                                            int stride, int lane) {
  int l = lane & 15, hi = lane >> 4;
  const _Float16* p = ldsT + (col0 + l) * stride + k0 + hi * 8;
  v8h a = *(const v8h*)p;
  v8h b = *(const v8h*)(p + 16);
  return __builtin_shufflevector(a, b, 0,1,2,3,4,5,6,7,8,9,10,11,12,13,14,15);
}

__device__ __forceinline__ void copy_lds(void* dst, const void* src, int bytes,
                                         int tid, int nthreads) {
  unsigned* d = (unsigned*)dst;
  const unsigned* s = (const unsigned*)src;
  for (int i = tid; i < bytes / 4; i += nthreads) d[i] = s[i];
}

// Gather edge features into hA[64][LDA] f16: cols 0..63 = x[nbr], 64..66 = p-q, rest 0
__device__ __forceinline__ void build_hA(_Float16* hA, const float* x, const float* pos,
                                         const int* nbr, const float* qbuf,
                                         int g, int b, int tid) {
  int r = tid >> 1;           // row (edge slot) 0..63, two threads per row
  int half = tid & 1;
  int nidx = nbr[(size_t)g * Kc + r];
  const float* xrow = x + ((size_t)b * Nc + nidx) * Fc;
  const float* prow = pos + ((size_t)b * Nc + nidx) * 3;
  float qx = qbuf[g * 3 + 0], qy = qbuf[g * 3 + 1], qz = qbuf[g * 3 + 2];
  int c0 = half * 52;
  for (int j = 0; j < 52; j++) {
    int c = c0 + j;
    float v;
    if (c < 64)       v = xrow[c];
    else if (c == 64) v = prow[0] - qx;
    else if (c == 65) v = prow[1] - qy;
    else if (c == 66) v = prow[2] - qz;
    else              v = 0.f;
    hA[r * LDA + c] = (_Float16)v;
  }
}

// Layer 1: (64x96)*(96x64) per group. Wave w owns row-tile w; 4 N-tiles, K=3 steps.
__device__ __forceinline__ void gemm1_compute(const _Float16* hA, const _Float16* w1t,
                                              const float* b1g, int wave, int lane,
                                              v8f c[4]) {
  for (int n = 0; n < 4; n++) { v8f z = {}; c[n] = z; }
#pragma unroll
  for (int k0 = 0; k0 < 96; k0 += 32) {
    v16h a = load_frag_A(hA, wave * 16, k0, LDA, lane);
#pragma unroll
    for (int n = 0; n < 4; n++) {
      v16h b = load_frag_B(w1t, n * 16, k0, LDA, lane);
      c[n] = __builtin_amdgcn_wmma_f32_16x16x32_f16(false, a, false, b,
                                                    (short)0, c[n], false, false);
    }
  }
  int l15 = lane & 15;
#pragma unroll
  for (int n = 0; n < 4; n++) {
    float bias = b1g[n * 16 + l15];
#pragma unroll
    for (int i = 0; i < 8; i++) {
      float v = c[n][i] + bias;
      c[n][i] = v > 0.f ? v : 0.f;
    }
  }
}

// Layer 2: (64x64)*(64x128). 8 N-tiles, K=2 steps.
__device__ __forceinline__ void gemm2_compute(const _Float16* hmid, const _Float16* w2t,
                                              const float* b2g, int wave, int lane,
                                              v8f d[8]) {
  for (int n = 0; n < 8; n++) { v8f z = {}; d[n] = z; }
#pragma unroll
  for (int k0 = 0; k0 < 64; k0 += 32) {
    v16h a = load_frag_A(hmid, wave * 16, k0, LDB, lane);
#pragma unroll
    for (int n = 0; n < 8; n++) {
      v16h b = load_frag_B(w2t, n * 16, k0, LDB, lane);
      d[n] = __builtin_amdgcn_wmma_f32_16x16x32_f16(false, a, false, b,
                                                    (short)0, d[n], false, false);
    }
  }
  int l15 = lane & 15;
#pragma unroll
  for (int n = 0; n < 8; n++) {
    float bias = b2g[n * 16 + l15];
#pragma unroll
    for (int i = 0; i < 8; i++) {
      float v = d[n][i] + bias;
      d[n][i] = v > 0.f ? v : 0.f;
    }
  }
}

// ---------------- K0: weight transpose/pad to f16, zero counters ------------
__global__ void prep_kernel(const float* __restrict__ W1, const float* __restrict__ W2,
                            _Float16* __restrict__ w1t, _Float16* __restrict__ w2t,
                            int* __restrict__ cntTotal) {
  int t = threadIdx.x;
  if (t == 0) *cntTotal = 0;
  for (int i = t; i < 64 * LDA; i += 256) {
    int col = i / LDA, k = i % LDA;
    w1t[i] = (_Float16)((k < 67) ? W1[k * 64 + col] : 0.f);
  }
  for (int i = t; i < 128 * LDB; i += 256) {
    int col = i / LDB, k = i % LDB;
    w2t[i] = (_Float16)((k < 64) ? W2[k * 128 + col] : 0.f);
  }
}

// ---------------- K1: farthest point sampling (1 block / cloud) -------------
__global__ void __launch_bounds__(256) fps_kernel(const float* __restrict__ pos,
                                                  int* __restrict__ idx_out) {
  int b = blockIdx.x, t = threadIdx.x;
  __shared__ float bpt[3];
  __shared__ float redv[8];
  __shared__ int   redi[8];
  __shared__ int   nxts;
  const float* base = pos + (size_t)b * Nc * 3;
  float px[8], py[8], pz[8], mind[8];
  for (int j = 0; j < 8; j++) {
    int i = t + 256 * j;
    px[j] = base[i * 3 + 0]; py[j] = base[i * 3 + 1]; pz[j] = base[i * 3 + 2];
  }
  if (t == 0) { bpt[0] = px[0]; bpt[1] = py[0]; bpt[2] = pz[0]; idx_out[b * Mc] = 0; }
  __syncthreads();
  float qx = bpt[0], qy = bpt[1], qz = bpt[2];
  for (int j = 0; j < 8; j++) {
    float dx = px[j] - qx, dy = py[j] - qy, dz = pz[j] - qz;
    mind[j] = dx * dx + dy * dy + dz * dz;
  }
  int lane = t & 31, wid = t >> 5;
  for (int s = 1; s < Mc; s++) {
    float bv = -1.f; int bi = 0;
    for (int j = 0; j < 8; j++) {               // ascending j -> keeps first max
      if (mind[j] > bv) { bv = mind[j]; bi = t + 256 * j; }
    }
    for (int off = 16; off > 0; off >>= 1) {    // argmax, tie -> smaller index
      float ov = __shfl_xor(bv, off, 32);
      int   oi = __shfl_xor(bi, off, 32);
      if (ov > bv || (ov == bv && oi < bi)) { bv = ov; bi = oi; }
    }
    if (lane == 0) { redv[wid] = bv; redi[wid] = bi; }
    __syncthreads();
    if (t == 0) {
      float fv = redv[0]; int fi = redi[0];
      for (int w = 1; w < 8; w++)
        if (redv[w] > fv || (redv[w] == fv && redi[w] < fi)) { fv = redv[w]; fi = redi[w]; }
      nxts = fi;
      idx_out[b * Mc + s] = fi;
    }
    __syncthreads();
    int nxt = nxts;
    if ((nxt & 255) == t) {
      int j = nxt >> 8;
      bpt[0] = px[j]; bpt[1] = py[j]; bpt[2] = pz[j];
    }
    __syncthreads();
    qx = bpt[0]; qy = bpt[1]; qz = bpt[2];
    for (int j = 0; j < 8; j++) {
      float dx = px[j] - qx, dy = py[j] - qy, dz = pz[j] - qz;
      float d = dx * dx + dy * dy + dz * dz;
      mind[j] = fminf(mind[j], d);
    }
  }
}

// ---------------- K2: ball query (1 wave per center), + misc outputs --------
__global__ void __launch_bounds__(256) ballquery_kernel(
    const float* __restrict__ pos, const int* __restrict__ idx_loc,
    int* __restrict__ nbr, int* __restrict__ vcnt, float* __restrict__ qbuf,
    int* __restrict__ cntTotal, float* __restrict__ pos_out,
    int* __restrict__ batch_out, int* __restrict__ idxg_out) {
  int g = blockIdx.x * 8 + (threadIdx.x >> 5);
  int lane = threadIdx.x & 31;
  int b = g >> 10;
  int ci = idx_loc[g];
  const float* base = pos + (size_t)b * Nc * 3;
  float qx = base[ci * 3 + 0], qy = base[ci * 3 + 1], qz = base[ci * 3 + 2];
  int cnt = 0;
  for (int s0 = 0; s0 < Nc && cnt < Kc; s0 += 32) {
    int i = s0 + lane;
    float dx = base[i * 3 + 0] - qx, dy = base[i * 3 + 1] - qy, dz = base[i * 3 + 2] - qz;
    bool in = (dx * dx + dy * dy + dz * dz) <= R2c;
    unsigned bal = __builtin_amdgcn_ballot_w32(in);
    int before = __popc(bal & ((1u << lane) - 1u));
    int slot = cnt + before;
    if (in && slot < Kc) nbr[(size_t)g * Kc + slot] = i;
    cnt += __popc(bal);
  }
  int vc = cnt < Kc ? cnt : Kc;
  for (int slot = vc + lane; slot < Kc; slot += 32) nbr[(size_t)g * Kc + slot] = ci;
  if (lane == 0) {
    vcnt[g] = vc;
    atomicAdd(cntTotal, vc);
    qbuf[g * 3 + 0] = qx; qbuf[g * 3 + 1] = qy; qbuf[g * 3 + 2] = qz;
    pos_out[g * 3 + 0] = qx; pos_out[g * 3 + 1] = qy; pos_out[g * 3 + 2] = qz;
    batch_out[g] = b;
    idxg_out[g] = b * Nc + ci;
  }
}

// ---------------- K3: pass 1 — GEMM1 + masked stats -------------------------
__global__ void __launch_bounds__(128) pass1_kernel(
    const float* __restrict__ x, const float* __restrict__ pos,
    const int* __restrict__ nbr, const int* __restrict__ vcnt,
    const float* __restrict__ qbuf, const _Float16* __restrict__ w1t_g,
    const float* __restrict__ b1, float* __restrict__ partial1) {
  __shared__ __attribute__((aligned(16))) _Float16 hA[64 * LDA];
  __shared__ __attribute__((aligned(16))) _Float16 w1t[64 * LDA];
  __shared__ float bsum[4][64], bsq[4][64];
  int g = blockIdx.x, b = g >> 10, tid = threadIdx.x;
  copy_lds(w1t, w1t_g, 64 * LDA * 2, tid, 128);
  build_hA(hA, x, pos, nbr, qbuf, g, b, tid);
  __syncthreads();
  int wave = tid >> 5, lane = tid & 31, l15 = lane & 15, hi = lane >> 4;
  v8f c[4];
  gemm1_compute(hA, w1t, b1, wave, lane, c);
  int vc = vcnt[g];
#pragma unroll
  for (int n = 0; n < 4; n++) {
    float s = 0.f, q = 0.f;
#pragma unroll
    for (int i = 0; i < 8; i++) {
      int row = wave * 16 + i + hi * 8;
      if (row < vc) { float v = c[n][i]; s += v; q += v * v; }
    }
    s += __shfl_xor(s, 16, 32);
    q += __shfl_xor(q, 16, 32);
    if (hi == 0) { bsum[wave][n * 16 + l15] = s; bsq[wave][n * 16 + l15] = q; }
  }
  __syncthreads();
  if (tid < 64) {
    partial1[(size_t)g * 128 + tid] =
        bsum[0][tid] + bsum[1][tid] + bsum[2][tid] + bsum[3][tid];
  } else {
    int cc = tid - 64;
    partial1[(size_t)g * 128 + 64 + cc] =
        bsq[0][cc] + bsq[1][cc] + bsq[2][cc] + bsq[3][cc];
  }
}

// ---------------- K4a/K6a: column-wise reduction of block partials ----------
__global__ void __launch_bounds__(256) reduce_kernel(const float* __restrict__ partial,
                                                     float* __restrict__ tot,
                                                     int stride, int ngroups) {
  int c = blockIdx.x, t = threadIdx.x;
  float s = 0.f;
  for (int g = t; g < ngroups; g += 256) s += partial[(size_t)g * stride + c];
  __shared__ float red[256];
  red[t] = s;
  __syncthreads();
  for (int off = 128; off > 0; off >>= 1) {
    if (t < off) red[t] += red[t + off];
    __syncthreads();
  }
  if (t == 0) tot[c] = red[0];
}

// ---------------- K4b/K6b: fold mean/var/gamma/beta -> scale/shift ----------
__global__ void finalize_kernel(const float* __restrict__ tot,
                                const float* __restrict__ gamma,
                                const float* __restrict__ beta,
                                const int* __restrict__ cntp,
                                float* __restrict__ scale, float* __restrict__ shift,
                                int C) {
  int c = threadIdx.x;
  if (c >= C) return;
  float cnt = (float)(*cntp);
  float mean = tot[c] / cnt;
  float var = tot[C + c] / cnt - mean * mean;
  float s = gamma[c] * rsqrtf(var + EPSc);
  scale[c] = s;
  shift[c] = beta[c] - mean * s;
}

// ---------------- K5: pass 2 — GEMM1+BN1+GEMM2 + masked stats ---------------
__global__ void __launch_bounds__(128) pass2_kernel(
    const float* __restrict__ x, const float* __restrict__ pos,
    const int* __restrict__ nbr, const int* __restrict__ vcnt,
    const float* __restrict__ qbuf, const _Float16* __restrict__ w1t_g,
    const _Float16* __restrict__ w2t_g, const float* __restrict__ b1,
    const float* __restrict__ b2, const float* __restrict__ scale1,
    const float* __restrict__ shift1, float* __restrict__ partial2) {
  __shared__ __attribute__((aligned(16))) _Float16 hA[64 * LDA];
  __shared__ __attribute__((aligned(16))) _Float16 w1t[64 * LDA];
  __shared__ __attribute__((aligned(16))) _Float16 w2t[128 * LDB];
  __shared__ __attribute__((aligned(16))) _Float16 hmid[64 * LDB];
  __shared__ float bsum[4][128], bsq[4][128];
  int g = blockIdx.x, b = g >> 10, tid = threadIdx.x;
  copy_lds(w1t, w1t_g, 64 * LDA * 2, tid, 128);
  copy_lds(w2t, w2t_g, 128 * LDB * 2, tid, 128);
  build_hA(hA, x, pos, nbr, qbuf, g, b, tid);
  for (int i = tid; i < 64 * 8; i += 128)  // zero hmid K-pad cols 64..71
    hmid[(i >> 3) * LDB + 64 + (i & 7)] = (_Float16)0.f;
  __syncthreads();
  int wave = tid >> 5, lane = tid & 31, l15 = lane & 15, hi = lane >> 4;
  v8f c[4];
  gemm1_compute(hA, w1t, b1, wave, lane, c);
#pragma unroll
  for (int n = 0; n < 4; n++) {
    int ch = n * 16 + l15;
    float sc = scale1[ch], sh = shift1[ch];
#pragma unroll
    for (int i = 0; i < 8; i++) {
      int row = wave * 16 + i + hi * 8;
      hmid[row * LDB + ch] = (_Float16)(c[n][i] * sc + sh);
    }
  }
  __syncthreads();
  v8f d[8];
  gemm2_compute(hmid, w2t, b2, wave, lane, d);
  int vc = vcnt[g];
#pragma unroll
  for (int n = 0; n < 8; n++) {
    float s = 0.f, q = 0.f;
#pragma unroll
    for (int i = 0; i < 8; i++) {
      int row = wave * 16 + i + hi * 8;
      if (row < vc) { float v = d[n][i]; s += v; q += v * v; }
    }
    s += __shfl_xor(s, 16, 32);
    q += __shfl_xor(q, 16, 32);
    if (hi == 0) { bsum[wave][n * 16 + l15] = s; bsq[wave][n * 16 + l15] = q; }
  }
  __syncthreads();
  if (tid < 128) {
    partial2[(size_t)g * 256 + tid] =
        bsum[0][tid] + bsum[1][tid] + bsum[2][tid] + bsum[3][tid];
    partial2[(size_t)g * 256 + 128 + tid] =
        bsq[0][tid] + bsq[1][tid] + bsq[2][tid] + bsq[3][tid];
  }
}

// ---------------- K7: pass 3 — full recompute + BN2 + masked max ------------
__global__ void __launch_bounds__(128) pass3_kernel(
    const float* __restrict__ x, const float* __restrict__ pos,
    const int* __restrict__ nbr, const int* __restrict__ vcnt,
    const float* __restrict__ qbuf, const _Float16* __restrict__ w1t_g,
    const _Float16* __restrict__ w2t_g, const float* __restrict__ b1,
    const float* __restrict__ b2, const float* __restrict__ scale1,
    const float* __restrict__ shift1, const float* __restrict__ scale2,
    const float* __restrict__ shift2, float* __restrict__ xout) {
  __shared__ __attribute__((aligned(16))) _Float16 hA[64 * LDA];
  __shared__ __attribute__((aligned(16))) _Float16 w1t[64 * LDA];
  __shared__ __attribute__((aligned(16))) _Float16 w2t[128 * LDB];
  __shared__ __attribute__((aligned(16))) _Float16 hmid[64 * LDB];
  __shared__ float wavemax[4][128];
  int g = blockIdx.x, b = g >> 10, tid = threadIdx.x;
  copy_lds(w1t, w1t_g, 64 * LDA * 2, tid, 128);
  copy_lds(w2t, w2t_g, 128 * LDB * 2, tid, 128);
  build_hA(hA, x, pos, nbr, qbuf, g, b, tid);
  for (int i = tid; i < 64 * 8; i += 128)
    hmid[(i >> 3) * LDB + 64 + (i & 7)] = (_Float16)0.f;
  __syncthreads();
  int wave = tid >> 5, lane = tid & 31, l15 = lane & 15, hi = lane >> 4;
  v8f c[4];
  gemm1_compute(hA, w1t, b1, wave, lane, c);
#pragma unroll
  for (int n = 0; n < 4; n++) {
    int ch = n * 16 + l15;
    float sc = scale1[ch], sh = shift1[ch];
#pragma unroll
    for (int i = 0; i < 8; i++) {
      int row = wave * 16 + i + hi * 8;
      hmid[row * LDB + ch] = (_Float16)(c[n][i] * sc + sh);
    }
  }
  __syncthreads();
  v8f d[8];
  gemm2_compute(hmid, w2t, b2, wave, lane, d);
  int vc = vcnt[g];
  const float NEG_INF = -__builtin_huge_valf();
#pragma unroll
  for (int n = 0; n < 8; n++) {
    int ch = n * 16 + l15;
    float sc = scale2[ch], sh = shift2[ch];
    float mx = NEG_INF;
#pragma unroll
    for (int i = 0; i < 8; i++) {
      int row = wave * 16 + i + hi * 8;
      if (row < vc) mx = fmaxf(mx, d[n][i] * sc + sh);
    }
    mx = fmaxf(mx, __shfl_xor(mx, 16, 32));
    if (hi == 0) wavemax[wave][ch] = mx;
  }
  __syncthreads();
  if (tid < 128) {
    float r = fmaxf(fmaxf(wavemax[0][tid], wavemax[1][tid]),
                    fmaxf(wavemax[2][tid], wavemax[3][tid]));
    xout[(size_t)g * C2c + tid] = r;
  }
}

// ---------------------------------------------------------------------------
extern "C" void kernel_launch(void* const* d_in, const int* in_sizes, int n_in,
                              void* d_out, int out_size, void* d_ws, size_t ws_size,
                              hipStream_t stream) {
  (void)in_sizes; (void)n_in; (void)out_size; (void)ws_size;
  const float* x   = (const float*)d_in[0];
  const float* pos = (const float*)d_in[1];
  const float* W1  = (const float*)d_in[3];
  const float* b1  = (const float*)d_in[4];
  const float* g1  = (const float*)d_in[5];
  const float* be1 = (const float*)d_in[6];
  const float* W2  = (const float*)d_in[7];
  const float* b2  = (const float*)d_in[8];
  const float* g2  = (const float*)d_in[9];
  const float* be2 = (const float*)d_in[10];

  // workspace carve-up (256B aligned sections)
  char* ws = (char*)d_ws;
  size_t o = 0;
  auto take = [&](size_t bytes) { size_t r = o; o = (o + bytes + 255) & ~(size_t)255; return r; };
  int*      idx_loc  = (int*)     (ws + take((size_t)NG * 4));
  int*      nbr      = (int*)     (ws + take((size_t)NG * Kc * 4));
  int*      vcnt     = (int*)     (ws + take((size_t)NG * 4));
  float*    qbuf     = (float*)   (ws + take((size_t)NG * 3 * 4));
  int*      cntTotal = (int*)     (ws + take(16));
  _Float16* w1t      = (_Float16*)(ws + take(64 * LDA * 2));
  _Float16* w2t      = (_Float16*)(ws + take(128 * LDB * 2));
  float*    tot1     = (float*)   (ws + take(128 * 4));
  float*    tot2     = (float*)   (ws + take(256 * 4));
  float*    sc1      = (float*)   (ws + take(64 * 4));
  float*    sh1      = (float*)   (ws + take(64 * 4));
  float*    sc2      = (float*)   (ws + take(128 * 4));
  float*    sh2      = (float*)   (ws + take(128 * 4));
  float*    partial1 = (float*)   (ws + take((size_t)NG * 128 * 4));
  float*    partial2 = (float*)   (ws + take((size_t)NG * 256 * 4));

  // output sections: x_out[NG*128] f32, pos_out[NG*3] f32, batch[NG] i32, idx[NG] i32
  float* xout = (float*)d_out;
  float* pout = xout + (size_t)NG * C2c;
  int*   bout = (int*)(pout + (size_t)NG * 3);
  int*   iout = bout + NG;

  prep_kernel<<<1, 256, 0, stream>>>(W1, W2, w1t, w2t, cntTotal);
  fps_kernel<<<Bc, 256, 0, stream>>>(pos, idx_loc);
  ballquery_kernel<<<NG / 8, 256, 0, stream>>>(pos, idx_loc, nbr, vcnt, qbuf,
                                               cntTotal, pout, bout, iout);
  pass1_kernel<<<NG, 128, 0, stream>>>(x, pos, nbr, vcnt, qbuf, w1t, b1, partial1);
  reduce_kernel<<<128, 256, 0, stream>>>(partial1, tot1, 128, NG);
  finalize_kernel<<<1, 64, 0, stream>>>(tot1, g1, be1, cntTotal, sc1, sh1, 64);
  pass2_kernel<<<NG, 128, 0, stream>>>(x, pos, nbr, vcnt, qbuf, w1t, w2t, b1, b2,
                                       sc1, sh1, partial2);
  reduce_kernel<<<256, 256, 0, stream>>>(partial2, tot2, 256, NG);
  finalize_kernel<<<1, 128, 0, stream>>>(tot2, g2, be2, cntTotal, sc2, sh2, 128);
  pass3_kernel<<<NG, 128, 0, stream>>>(x, pos, nbr, vcnt, qbuf, w1t, w2t, b1, b2,
                                       sc1, sh1, sc2, sh2, xout);
}
